// DotAttention_3530463117934
// MI455X (gfx1250) — matrix operand order repaired and checked by reading, not compiled, not run on hardware
//
#include <hip/hip_runtime.h>
#include <hip/hip_bf16.h>

// ---------------------------------------------------------------------------
// Problem constants (from reference): B=16, L=2048 (both axes), D=512, H=256
// ---------------------------------------------------------------------------
#define BATCH   16
#define L_SEQ   2048
#define D_DIM   512
#define H_DIM   256
#define NEG_INF (-1.0e30f)
#define SCALE   (0.0625f)   // 1/sqrt(256)

typedef __attribute__((ext_vector_type(16))) __bf16 v16bf;
typedef __attribute__((ext_vector_type(8)))  __bf16 v8bf;
typedef __attribute__((ext_vector_type(8)))  float  v8f;

// ---------------------------------------------------------------------------
// WMMA helper: D = A(16x32 bf16) * B(32x16 bf16) + C(16x16 f32)
// ---------------------------------------------------------------------------
__device__ __forceinline__ v8f wmma_bf16(v16bf a, v16bf b, v8f c) {
    return __builtin_amdgcn_wmma_f32_16x16x32_bf16(
        /*neg_a=*/false, a, /*neg_b=*/false, b,
        /*c_mod=*/(short)0, c, /*reuse_a=*/false, /*reuse_b=*/false);
}

// Build a 16-element bf16 fragment from two 16-byte runs (global or LDS).
template <typename P>
__device__ __forceinline__ v16bf load_frag16(const P* p0, const P* p1) {
    v8bf lo = *(const v8bf*)p0;
    v8bf hi = *(const v8bf*)p1;
    return __builtin_shufflevector(lo, hi,
        0, 1, 2, 3, 4, 5, 6, 7, 8, 9, 10, 11, 12, 13, 14, 15);
}

// A-fragment built from f32 memory with on-the-fly bf16 conversion.
__device__ __forceinline__ v16bf cvt_frag16(const float* p0, const float* p1) {
    v16bf r;
#pragma unroll
    for (int i = 0; i < 8; ++i) {
        r[i]     = (__bf16)p0[i];
        r[i + 8] = (__bf16)p1[i];
    }
    return r;
}

// ---------------------------------------------------------------------------
// Kernel 0: transpose + f32->bf16 convert.  src: [R][C] f32 -> dst: [C][R] bf16
// Used for W_in/W_mem (-> Wt[H][D]) and memory (-> Vt[b][d][l]).
// ---------------------------------------------------------------------------
__global__ __launch_bounds__(256) void transpose_cvt_kernel(
    const float* __restrict__ src, __bf16* __restrict__ dst,
    int R, int C, long srcBatchStride, long dstBatchStride)
{
    __shared__ float tile[32][33];
    const float* s = src + (size_t)blockIdx.z * srcBatchStride;
    __bf16*      d = dst + (size_t)blockIdx.z * dstBatchStride;

    const int r0 = blockIdx.x * 32;
    const int c0 = blockIdx.y * 32;
    const int tx = threadIdx.x;   // 0..31
    const int ty = threadIdx.y;   // 0..7

#pragma unroll
    for (int i = ty; i < 32; i += 8)
        tile[i][tx] = s[(size_t)(r0 + i) * C + c0 + tx];
    __syncthreads();
#pragma unroll
    for (int i = ty; i < 32; i += 8)
        d[(size_t)(c0 + i) * R + r0 + tx] = (__bf16)tile[tx][i];
}

// ---------------------------------------------------------------------------
// Kernel 1: Y = relu(X @ W) in bf16.   X: [R][512] f32 (R = B*L),
// Wt: [256][512] bf16 (transposed W), Y: [R][256] bf16.
// One workgroup = 16 rows; 8 waves each own a 32-wide slice of H.
// ---------------------------------------------------------------------------
__global__ __launch_bounds__(256) void proj_relu_kernel(
    const float* __restrict__ X, const __bf16* __restrict__ Wt,
    __bf16* __restrict__ Y)
{
    const int lane = threadIdx.x & 31;
    const int wave = threadIdx.x >> 5;
    const int g    = lane >> 4;      // lane group (0: lanes 0-15, 1: 16-31)
    const int nl   = lane & 15;
    const int row0 = blockIdx.x * 16;
    const int nb   = wave * 32;      // this wave's H-column base

    v8f acc0 = {}, acc1 = {};
#pragma unroll
    for (int t = 0; t < 16; ++t) {                 // K loop: 512 / 32
        const float* ap = X + (size_t)(row0 + nl) * D_DIM + 32 * t + g * 16;
        v16bf a = cvt_frag16(ap, ap + 8);
        const __bf16* b0 = Wt + (size_t)(nb + nl)      * D_DIM + 32 * t + g * 16;
        const __bf16* b1 = Wt + (size_t)(nb + 16 + nl) * D_DIM + 32 * t + g * 16;
        acc0 = wmma_bf16(a, load_frag16(b0, b0 + 8), acc0);
        acc1 = wmma_bf16(a, load_frag16(b1, b1 + 8), acc1);
    }
#pragma unroll
    for (int r = 0; r < 8; ++r) {
        const size_t row = (size_t)(row0 + r + 8 * g);
        Y[row * H_DIM + nb + nl]      = (__bf16)fmaxf(acc0[r], 0.0f);
        Y[row * H_DIM + nb + 16 + nl] = (__bf16)fmaxf(acc1[r], 0.0f);
    }
}

// ---------------------------------------------------------------------------
// Kernel 2: flash attention.
//   Q  : [B*L][256] bf16 (relu(inputs@W_in))
//   Kh : [B*L][256] bf16 (relu(memory@W_mem))
//   Vt : [B][512][L]  bf16 (transposed original memory)
//   out: [B][L][1024] f32  — this kernel writes cols [0,512)
// Workgroup: 16 query rows, 8 waves. Outer loop: 256 memory rows per step.
// Wave w: computes S for mem cols [32w,32w+32); owns D-slice [64w,64w+64).
// ---------------------------------------------------------------------------
__global__ __launch_bounds__(256) void attn_kernel(
    const __bf16* __restrict__ Q, const __bf16* __restrict__ Kh,
    const __bf16* __restrict__ Vt, const int* __restrict__ mask,
    float* __restrict__ out)
{
    __shared__ __bf16 Pl[16][H_DIM];   // probabilities tile (16 x 256), 8 KB
    __shared__ float  lmax[16][8];     // per-row local max, per wave
    __shared__ float  lsum[16][8];     // per-row local sum, per wave

    const int lane = threadIdx.x & 31;
    const int wave = threadIdx.x >> 5;
    const int g    = lane >> 4;
    const int nl   = lane & 15;
    const int b    = blockIdx.y;
    const int q0   = blockIdx.x * 16;

    const __bf16* Qb = Q  + ((size_t)b * L_SEQ + q0) * H_DIM;
    const __bf16* Kb = Kh + (size_t)b * L_SEQ * H_DIM;
    const __bf16* Vb = Vt + (size_t)b * D_DIM * L_SEQ;
    const int*    mb = mask + b * L_SEQ;

    // Preload Q fragments (16x256 -> 8 A-fragments of 16x32) into registers.
    v16bf qf[8];
#pragma unroll
    for (int t = 0; t < 8; ++t) {
        const __bf16* ap = Qb + (size_t)nl * H_DIM + 32 * t + g * 16;
        qf[t] = load_frag16(ap, ap + 8);
    }

    v8f o[4] = {};                 // 16x64 f32 context accumulator
    float Mrun[8], Lrun[8];
#pragma unroll
    for (int r = 0; r < 8; ++r) { Mrun[r] = -__builtin_inff(); Lrun[r] = 0.0f; }

    for (int j0 = 0; j0 < L_SEQ; j0 += 256) {
        const int jw = j0 + wave * 32;     // this wave's score-chunk base

        // ---- S = Qh @ Kh^T for this wave's 16x32 chunk (2 tiles) ----
        v8f s0 = {}, s1 = {};
#pragma unroll
        for (int t = 0; t < 8; ++t) {      // K loop over H = 256
            const __bf16* k0 = Kb + (size_t)(jw + nl)      * H_DIM + 32 * t + g * 16;
            const __bf16* k1 = Kb + (size_t)(jw + 16 + nl) * H_DIM + 32 * t + g * 16;
            s0 = wmma_bf16(qf[t], load_frag16(k0, k0 + 8), s0);
            s1 = wmma_bf16(qf[t], load_frag16(k1, k1 + 8), s1);
        }
        if (j0 + 256 < L_SEQ) {            // gfx1250 prefetch of next K slab
            __builtin_prefetch(Kb + (size_t)(jw + 256 + nl) * H_DIM, 0, 3);
        }

        // ---- scale + mask (each lane holds one memory column per tile) ----
        const bool m0 = mb[jw + nl] != 0;
        const bool m1 = mb[jw + 16 + nl] != 0;
#pragma unroll
        for (int r = 0; r < 8; ++r) {
            s0[r] = m0 ? s0[r] * SCALE : NEG_INF;
            s1[r] = m1 ? s1[r] * SCALE : NEG_INF;
        }

        // ---- local row max over the 32 columns of this wave's chunk ----
        float mx[8];
#pragma unroll
        for (int r = 0; r < 8; ++r) mx[r] = fmaxf(s0[r], s1[r]);
#pragma unroll
        for (int off = 1; off < 16; off <<= 1)
#pragma unroll
            for (int r = 0; r < 8; ++r)
                mx[r] = fmaxf(mx[r], __shfl_xor(mx[r], off, 16));
        if (nl == 0) {
#pragma unroll
            for (int r = 0; r < 8; ++r) lmax[r + 8 * g][wave] = mx[r];
        }
        __syncthreads();

        // ---- global (all-wave + running) max per row ----
        float gm[8];
#pragma unroll
        for (int r = 0; r < 8; ++r) {
            float v = Mrun[r];
#pragma unroll
            for (int w = 0; w < 8; ++w) v = fmaxf(v, lmax[r + 8 * g][w]);
            gm[r] = v;
        }

        // ---- P = exp(S - gm), write to LDS; local row sums ----
        float sm[8];
#pragma unroll
        for (int r = 0; r < 8; ++r) {
            const float p0 = __expf(s0[r] - gm[r]);
            const float p1 = __expf(s1[r] - gm[r]);
            sm[r] = p0 + p1;
            Pl[r + 8 * g][wave * 32 + nl]      = (__bf16)p0;
            Pl[r + 8 * g][wave * 32 + 16 + nl] = (__bf16)p1;
        }
#pragma unroll
        for (int off = 1; off < 16; off <<= 1)
#pragma unroll
            for (int r = 0; r < 8; ++r)
                sm[r] += __shfl_xor(sm[r], off, 16);
        if (nl == 0) {
#pragma unroll
            for (int r = 0; r < 8; ++r) lsum[r + 8 * g][wave] = sm[r];
        }
        __syncthreads();

        // ---- online-softmax rescale of accumulator + running stats ----
        float al[8];
#pragma unroll
        for (int r = 0; r < 8; ++r) {
            al[r]   = __expf(Mrun[r] - gm[r]);
            Mrun[r] = gm[r];
            float ls = 0.0f;
#pragma unroll
            for (int w = 0; w < 8; ++w) ls += lsum[r + 8 * g][w];
            Lrun[r] = Lrun[r] * al[r] + ls;
        }
#pragma unroll
        for (int jt = 0; jt < 4; ++jt)
#pragma unroll
            for (int r = 0; r < 8; ++r) o[jt][r] *= al[r];

        // ---- O += P @ V over 256 memory rows; this wave's 64 D-columns ----
#pragma unroll
        for (int t = 0; t < 8; ++t) {
            const __bf16* pp = &Pl[nl][32 * t] + g * 16;
            v16bf pa = load_frag16(pp, pp + 8);
#pragma unroll
            for (int jt = 0; jt < 4; ++jt) {
                const int d = wave * 64 + jt * 16 + nl;
                const __bf16* vp = Vb + (size_t)d * L_SEQ + j0 + 32 * t + g * 16;
                o[jt] = wmma_bf16(pa, load_frag16(vp, vp + 8), o[jt]);
            }
        }
        // No extra barrier needed: next iteration's LDS writes are fenced by
        // its own __syncthreads() before any Pl/lsum producers run.
    }

    // ---- normalize and write context to out[:, 0:512] ----
#pragma unroll
    for (int r = 0; r < 8; ++r) {
        const float inv = 1.0f / Lrun[r];
        const size_t row = (size_t)b * L_SEQ + q0 + r + 8 * g;
#pragma unroll
        for (int jt = 0; jt < 4; ++jt)
            out[row * 1024 + wave * 64 + jt * 16 + nl] = o[jt][r] * inv;
    }
}

// ---------------------------------------------------------------------------
// Kernel 3: out[:, 512:1024] = inputs (vectorized float4 copy)
// ---------------------------------------------------------------------------
__global__ __launch_bounds__(256) void concat_copy_kernel(
    const float4* __restrict__ in, float4* __restrict__ out, int total4)
{
    const int i = blockIdx.x * blockDim.x + threadIdx.x;
    if (i < total4) {
        const int row = i >> 7;       // 128 float4 per 512-f32 input row
        const int col = i & 127;
        out[(size_t)row * 256 + 128 + col] = in[i];
    }
}

// ---------------------------------------------------------------------------
// Host launcher
// ---------------------------------------------------------------------------
extern "C" void kernel_launch(void* const* d_in, const int* in_sizes, int n_in,
                              void* d_out, int out_size, void* d_ws, size_t ws_size,
                              hipStream_t stream)
{
    (void)in_sizes; (void)n_in; (void)out_size; (void)ws_size;

    const float* inputs = (const float*)d_in[0];  // [16][2048][512]
    const float* memory = (const float*)d_in[1];  // [16][2048][512]
    const int*   mmask  = (const int*)  d_in[2];  // [16][2048]
    const float* W_in   = (const float*)d_in[3];  // [512][256]
    const float* W_mem  = (const float*)d_in[4];  // [512][256]
    float*       out    = (float*)d_out;          // [16][2048][1024]

    // Workspace layout (bf16):
    //   Q  [B*L][256]  = 16 MB
    //   Kh [B*L][256]  = 16 MB
    //   Vt [B][512][L] = 32 MB
    //   WtIn/WtMem [256][512] = 0.25 MB each
    __bf16* Q     = (__bf16*)d_ws;
    __bf16* Kh    = Q  + (size_t)BATCH * L_SEQ * H_DIM;
    __bf16* Vt    = Kh + (size_t)BATCH * L_SEQ * H_DIM;
    __bf16* WtIn  = Vt + (size_t)BATCH * D_DIM * L_SEQ;
    __bf16* WtMem = WtIn + (size_t)H_DIM * D_DIM;

    // Phase 0: transposed bf16 operands.
    {
        dim3 blk(32, 8);
        transpose_cvt_kernel<<<dim3(D_DIM / 32, H_DIM / 32, 1), blk, 0, stream>>>(
            W_in, WtIn, D_DIM, H_DIM, 0, 0);
        transpose_cvt_kernel<<<dim3(D_DIM / 32, H_DIM / 32, 1), blk, 0, stream>>>(
            W_mem, WtMem, D_DIM, H_DIM, 0, 0);
        transpose_cvt_kernel<<<dim3(L_SEQ / 32, D_DIM / 32, BATCH), blk, 0, stream>>>(
            memory, Vt, L_SEQ, D_DIM,
            (long)L_SEQ * D_DIM, (long)D_DIM * L_SEQ);
    }

    // Phase 1: projections (WMMA).
    {
        const int rowBlocks = BATCH * L_SEQ / 16;   // 2048
        proj_relu_kernel<<<rowBlocks, 256, 0, stream>>>(inputs, WtIn, Q);
        proj_relu_kernel<<<rowBlocks, 256, 0, stream>>>(memory, WtMem, Kh);
    }

    // Phase 2: fused masked-softmax attention (WMMA flash loop).
    attn_kernel<<<dim3(L_SEQ / 16, BATCH), 256, 0, stream>>>(Q, Kh, Vt, mmask, out);

    // Phase 3: concat inputs into out[:, 512:1024].
    {
        const int total4 = BATCH * L_SEQ * D_DIM / 4;
        concat_copy_kernel<<<(total4 + 255) / 256, 256, 0, stream>>>(
            (const float4*)inputs, (float4*)out, total4);
    }
}
